// VQAELight_58600533787125
// MI455X (gfx1250) — matrix operand order repaired and checked by reading, not compiled, not run on hardware
//
#include <hip/hip_runtime.h>

#define N_ROWS 32768
#define DIM    256
#define K_CB   4096
#define BT_STRIDE 272   // 256 + 16 bf16 pad to spread LDS banks
#define TILE_COLS 32
#define NT (K_CB / TILE_COLS)   // 128 column tiles

typedef __attribute__((ext_vector_type(16))) __bf16 v16bf;
typedef __attribute__((ext_vector_type(8)))  __bf16 v8bf;
typedef __attribute__((ext_vector_type(8)))  float  v8f;
typedef __attribute__((ext_vector_type(4)))  int    v4i;

typedef __attribute__((address_space(1))) v4i* gptr_v4i;
typedef __attribute__((address_space(3))) v4i* lptr_v4i;

__device__ __forceinline__ unsigned int pack_bf16x2(float a, float b) {
    unsigned int xa = __float_as_uint(a);
    unsigned int xb = __float_as_uint(b);
    xa = (xa + 0x7FFFu + ((xa >> 16) & 1u)) >> 16;
    xb = (xb + 0x7FFFu + ((xb >> 16) & 1u)) >> 16;
    return xa | (xb << 16);
}

// One wave normalizes one row of D=256 floats and stores bf16 (RNE).
__global__ void vq_normalize_bf16(const float* __restrict__ src,
                                  unsigned short* __restrict__ dst,
                                  int rows) {
    int gid  = blockIdx.x * blockDim.x + threadIdx.x;
    int wave = gid >> 5;
    int lane = gid & 31;
    if (wave >= rows) return;
    const float4* p = reinterpret_cast<const float4*>(src + (size_t)wave * DIM + lane * 8);
    float4 v0 = p[0], v1 = p[1];
    float s = v0.x*v0.x + v0.y*v0.y + v0.z*v0.z + v0.w*v0.w
            + v1.x*v1.x + v1.y*v1.y + v1.z*v1.z + v1.w*v1.w;
    #pragma unroll
    for (int m = 1; m < 32; m <<= 1) s += __shfl_xor(s, m, 32);
    float inv = 1.0f / fmaxf(sqrtf(s), 1e-12f);
    uint4 pk;
    pk.x = pack_bf16x2(v0.x * inv, v0.y * inv);
    pk.y = pack_bf16x2(v0.z * inv, v0.w * inv);
    pk.z = pack_bf16x2(v1.x * inv, v1.y * inv);
    pk.w = pack_bf16x2(v1.z * inv, v1.w * inv);
    *reinterpret_cast<uint4*>(dst + (size_t)wave * DIM + lane * 8) = pk;
}

// Issue the async-to-LDS copies for one 32-column codebook tile.
// 32 cols x 512B = 16 KB = 1024 x 16B chunks; 256 threads -> 4 chunks each.
__device__ __forceinline__ void stage_tile_async(const unsigned short* __restrict__ en,
                                                 unsigned short* lbuf, int ct, int tid) {
    const int sub   = tid & 31;   // 16B chunk within a row
    const int cbase = tid >> 5;   // 0..7
    #pragma unroll
    for (int r = 0; r < 4; ++r) {
        const int col = cbase + r * 8;
        const unsigned short* g = en + (size_t)(ct * TILE_COLS + col) * DIM + sub * 8;
        unsigned short* l = lbuf + col * BT_STRIDE + sub * 8;
        __builtin_amdgcn_global_load_async_to_lds_b128(
            (gptr_v4i)g, (lptr_v4i)l, 0, 0);
    }
}

// Fused  S = Xn * En^T  (bf16 WMMA, f32 acc)  +  per-row argmax epilogue.
// Block = 8 waves; wave w owns 16 rows; block sweeps all 4096 codes in
// 32-column tiles, double-buffered through LDS via async-to-LDS DMA.
__global__ void __launch_bounds__(256)
vq_gemm_argmax(const unsigned short* __restrict__ xn,
               const unsigned short* __restrict__ en,
               int* __restrict__ idx) {
    __shared__ unsigned short btile[2][TILE_COLS * BT_STRIDE];

    const int lane    = threadIdx.x & 31;
    const int waveId  = threadIdx.x >> 5;
    const int rowBase = (blockIdx.x * 8 + waveId) * 16;

    // ---- A fragments: 16 rows x 256 K of normalized inputs -> registers.
    // 16-bit A layout: lanes 0-15 row M=lane, K {0..7,16..23}; lanes 16-31 K {8..15,24..31}
    const size_t rOff = (size_t)(rowBase + (lane & 15)) * DIM;
    const int kA = (lane < 16) ? 0 : 8;
    v16bf afr[8];
    #pragma unroll
    for (int kk = 0; kk < 8; ++kk) {
        v8bf lo = *reinterpret_cast<const v8bf*>(xn + rOff + kk * 32 + kA);
        v8bf hi = *reinterpret_cast<const v8bf*>(xn + rOff + kk * 32 + kA + 16);
        afr[kk] = __builtin_shufflevector(lo, hi, 0,1,2,3,4,5,6,7,8,9,10,11,12,13,14,15);
    }

    float best[8];
    int   bcol[8];
    #pragma unroll
    for (int j = 0; j < 8; ++j) { best[j] = -3.4e38f; bcol[j] = 0; }

    // 16-bit B layout: lanes 0-15 col N=lane hold K {0..7,16..23}; lanes 16-31 hold K {8..15,24..31}
    const int kB = (lane < 16) ? 0 : 8;

    // prologue: kick off tile 0
    stage_tile_async(en, &btile[0][0], 0, threadIdx.x);

    for (int ct = 0; ct < NT; ++ct) {
        if (ct + 1 < NT) {
            // prefetch next tile into the other buffer, then wait until only
            // those 4 fresh loads remain outstanding (async completes in order)
            stage_tile_async(en, &btile[(ct + 1) & 1][0], ct + 1, threadIdx.x);
            asm volatile("s_wait_asynccnt 0x4" ::: "memory");
        } else {
            asm volatile("s_wait_asynccnt 0x0" ::: "memory");
        }
        __syncthreads();

        const unsigned short* b0 = &btile[ct & 1][0] + (lane & 15) * BT_STRIDE + kB;
        const unsigned short* b1 = b0 + 16 * BT_STRIDE;

        v8f acc0 = {0.f,0.f,0.f,0.f,0.f,0.f,0.f,0.f};
        v8f acc1 = {0.f,0.f,0.f,0.f,0.f,0.f,0.f,0.f};
        #pragma unroll
        for (int kk = 0; kk < 8; ++kk) {
            v8bf p0 = *reinterpret_cast<const v8bf*>(b0 + kk * 32);
            v8bf p1 = *reinterpret_cast<const v8bf*>(b0 + kk * 32 + 16);
            v8bf q0 = *reinterpret_cast<const v8bf*>(b1 + kk * 32);
            v8bf q1 = *reinterpret_cast<const v8bf*>(b1 + kk * 32 + 16);
            v16bf bf0 = __builtin_shufflevector(p0, p1, 0,1,2,3,4,5,6,7,8,9,10,11,12,13,14,15);
            v16bf bf1 = __builtin_shufflevector(q0, q1, 0,1,2,3,4,5,6,7,8,9,10,11,12,13,14,15);
            acc0 = __builtin_amdgcn_wmma_f32_16x16x32_bf16(
                false, afr[kk], false, bf0, (short)0, acc0, false, false);
            acc1 = __builtin_amdgcn_wmma_f32_16x16x32_bf16(
                false, afr[kk], false, bf1, (short)0, acc1, false, false);
        }

        const int col0 = ct * TILE_COLS + (lane & 15);
        #pragma unroll
        for (int j = 0; j < 8; ++j) {
            float v = acc0[j];                  // row M = j (+8 upper half), col N = lane&15
            if (v > best[j]) { best[j] = v; bcol[j] = col0; }
        }
        #pragma unroll
        for (int j = 0; j < 8; ++j) {
            float v = acc1[j];
            if (v > best[j]) { best[j] = v; bcol[j] = col0 + 16; }
        }
        __syncthreads();
    }

    // reduce argmax across the 16 lanes of each half-wave; lower index wins ties
    #pragma unroll
    for (int j = 0; j < 8; ++j) {
        float v = best[j];
        int   c = bcol[j];
        #pragma unroll
        for (int m = 1; m < 16; m <<= 1) {
            float ov = __shfl_xor(v, m, 32);
            int   oc = __shfl_xor(c, m, 32);
            if (ov > v || (ov == v && oc < c)) { v = ov; c = oc; }
        }
        if ((lane & 15) == 0) {
            int rowM = j + ((lane >> 4) << 3);
            idx[rowBase + rowM] = c;
        }
    }
}

// Gather un-normalized codes, write straight-through output, per-row sq-err, index-as-float.
__global__ void vq_gather_loss(const float* __restrict__ inp,
                               const float* __restrict__ emb,
                               const int* __restrict__ idx,
                               float* __restrict__ out,
                               float* __restrict__ partials) {
    __shared__ float sm[8];
    const int r = blockIdx.x;
    const int t = threadIdx.x;
    const int k = idx[r];
    float q = emb[(size_t)k * DIM + t];
    float x = inp[(size_t)r * DIM + t];
    out[(size_t)r * DIM + t] = q;          // flat + sg(quantized - flat) == quantized
    float d = q - x;
    float s = d * d;
    #pragma unroll
    for (int m = 1; m < 32; m <<= 1) s += __shfl_xor(s, m, 32);
    if ((t & 31) == 0) sm[t >> 5] = s;
    __syncthreads();
    if (t < 8) {
        float v = sm[t];
        #pragma unroll
        for (int m = 1; m < 8; m <<= 1) v += __shfl_xor(v, m, 32);
        if (t == 0) {
            partials[r] = v;
            out[(size_t)N_ROWS * DIM + 1 + r] = (float)k;   // indices (float dtype of d_out)
        }
    }
}

__global__ void vq_finalize(const float* __restrict__ partials,
                            float* __restrict__ out) {
    __shared__ float sm[8];
    const int t = threadIdx.x;
    float s = 0.f;
    for (int i = t; i < N_ROWS; i += 256) s += partials[i];
    #pragma unroll
    for (int m = 1; m < 32; m <<= 1) s += __shfl_xor(s, m, 32);
    if ((t & 31) == 0) sm[t >> 5] = s;
    __syncthreads();
    if (t < 8) {
        float v = sm[t];
        #pragma unroll
        for (int m = 1; m < 8; m <<= 1) v += __shfl_xor(v, m, 32);
        if (t == 0)
            out[(size_t)N_ROWS * DIM] = 0.25f * v / (float)((size_t)N_ROWS * DIM);
    }
}

extern "C" void kernel_launch(void* const* d_in, const int* in_sizes, int n_in,
                              void* d_out, int out_size, void* d_ws, size_t ws_size,
                              hipStream_t stream) {
    const float* inputs     = (const float*)d_in[0];   // [N, D] fp32
    const float* embeddings = (const float*)d_in[1];   // [K, D] fp32
    float* out = (float*)d_out;                        // [N*D | 1 | N]

    char* ws = (char*)d_ws;
    unsigned short* xn = (unsigned short*)ws;                                  // 16 MB
    unsigned short* en = (unsigned short*)(ws + (size_t)N_ROWS * DIM * 2);     //  2 MB
    int* idx      = (int*)  (ws + (size_t)N_ROWS * DIM * 2 + (size_t)K_CB * DIM * 2);
    float* parts  = (float*)(ws + (size_t)N_ROWS * DIM * 2 + (size_t)K_CB * DIM * 2
                                + (size_t)N_ROWS * sizeof(int));

    // 1) fp32 L2-normalize -> bf16 (one wave per row)
    vq_normalize_bf16<<<N_ROWS / 8, 256, 0, stream>>>(inputs, xn, N_ROWS);
    vq_normalize_bf16<<<K_CB   / 8, 256, 0, stream>>>(embeddings, en, K_CB);

    // 2) fused bf16-WMMA similarity + argmax (8 waves/block, 128 rows/block,
    //    async-to-LDS double-buffered codebook streaming)
    vq_gemm_argmax<<<N_ROWS / 128, 256, 0, stream>>>(xn, en, idx);

    // 3) gather + straight-through output + loss partials + indices
    vq_gather_loss<<<N_ROWS, 256, 0, stream>>>(inputs, embeddings, idx, out, parts);

    // 4) deterministic loss reduction
    vq_finalize<<<1, 256, 0, stream>>>(parts, out);
}
